// Multi_Head_Attention_2layer_5677946765802
// MI455X (gfx1250) — matrix operand order repaired and checked by reading, not compiled, not run on hardware
//
#include <hip/hip_runtime.h>
#include <hip/hip_bf16.h>

// ---------------------------------------------------------------------------
// Types for CDNA5 WMMA
// ---------------------------------------------------------------------------
typedef __attribute__((ext_vector_type(16))) __bf16 v16bf;
typedef __attribute__((ext_vector_type(8)))  float  v8f;

__device__ __forceinline__ unsigned short f2bf(float x) {
    unsigned int u = __builtin_bit_cast(unsigned int, x);
    u += 0x7FFFu + ((u >> 16) & 1u);           // round-to-nearest-even
    return (unsigned short)(u >> 16);
}

// load 4 consecutive elements -> raw bf16
__device__ __forceinline__ void load4_bf(const float* p, unsigned short o[4]) {
    float4 t = *(const float4*)p;
    o[0] = f2bf(t.x); o[1] = f2bf(t.y); o[2] = f2bf(t.z); o[3] = f2bf(t.w);
}
__device__ __forceinline__ void load4_bf(const unsigned short* p, unsigned short o[4]) {
    uint2 t = *(const uint2*)p;
    o[0] = (unsigned short)(t.x & 0xFFFFu); o[1] = (unsigned short)(t.x >> 16);
    o[2] = (unsigned short)(t.y & 0xFFFFu); o[3] = (unsigned short)(t.y >> 16);
}

// ---------------------------------------------------------------------------
// Tiled WMMA GEMM:  C[M,N] (f32) = A[M,K] * B + bias
//   AT: float or ushort(bf16 raw)      A row-major [M,K]
//   BT: float or ushort(bf16 raw)
//   B_PRETRANS=true : B given as [N,K]  (i.e. we compute A * B^T) -> direct copy
//   B_PRETRANS=false: B given as [K,N]  row-major -> transpose-scatter into LDS
// Block tile 128x128, K-stage 64 (two bf16 WMMA K-slices per LDS stage).
// 256 threads = 8 waves, wave grid 4(M) x 2(N), wave tile 32x64
// -> 16 x v_wmma_f32_16x16x32_bf16 per stage per wave.
// ---------------------------------------------------------------------------
template <typename AT, typename BT, bool B_PRETRANS, bool HAS_BIAS>
__global__ __launch_bounds__(256) void gemm_wmma(
    const AT* __restrict__ A, const BT* __restrict__ B,
    const float* __restrict__ bias, float* __restrict__ C,
    int M, int N, int K, int lda, int ldb, int ldc,
    size_t strideA, size_t strideB, size_t strideC)
{
    constexpr int BM = 128, BN = 128, BK = 64;
    constexpr int SA = BK + 8;                 // 72 elems -> 144B row stride (16B aligned)
    __shared__ unsigned short As[BM * SA];
    __shared__ unsigned short Bs[BN * SA];     // stored as B^T: [n][k]

    A += (size_t)blockIdx.z * strideA;
    B += (size_t)blockIdx.z * strideB;
    C += (size_t)blockIdx.z * strideC;

    const int tid  = threadIdx.x;
    const int lane = tid & 31;
    const int wid  = tid >> 5;
    const int wm   = wid >> 1;                 // 0..3
    const int wn   = wid & 1;                  // 0..1
    const int m0   = blockIdx.y * BM;
    const int n0   = blockIdx.x * BN;
    const int hf   = lane >> 4;                // half-wave select
    const int l15  = lane & 15;

    v8f acc[2][4];
    #pragma unroll
    for (int i = 0; i < 2; ++i)
        #pragma unroll
        for (int j = 0; j < 4; ++j)
            #pragma unroll
            for (int e = 0; e < 8; ++e) acc[i][j][e] = 0.0f;

    for (int kk = 0; kk < K; kk += BK) {
        __syncthreads();
        // ---- A tile: [128 rows][64 k] -> bf16 LDS
        #pragma unroll
        for (int it = 0; it < 8; ++it) {
            int cidx = tid + it * 256;         // 0..2047 chunks of 4
            int r    = cidx >> 4;              // 0..127
            int c4   = (cidx & 15) << 2;       // 0..60
            unsigned short o[4];
            load4_bf(A + (size_t)(m0 + r) * lda + kk + c4, o);
            uint2 w;
            w.x = (unsigned)o[0] | ((unsigned)o[1] << 16);
            w.y = (unsigned)o[2] | ((unsigned)o[3] << 16);
            *(uint2*)&As[r * SA + c4] = w;
        }
        // ---- B tile -> Bs as [n][k]
        if (B_PRETRANS) {
            #pragma unroll
            for (int it = 0; it < 8; ++it) {
                int cidx = tid + it * 256;
                int r    = cidx >> 4;          // n in tile
                int c4   = (cidx & 15) << 2;   // k in tile
                unsigned short o[4];
                load4_bf(B + (size_t)(n0 + r) * ldb + kk + c4, o);
                uint2 w;
                w.x = (unsigned)o[0] | ((unsigned)o[1] << 16);
                w.y = (unsigned)o[2] | ((unsigned)o[3] << 16);
                *(uint2*)&Bs[r * SA + c4] = w;
            }
        } else {
            #pragma unroll
            for (int it = 0; it < 8; ++it) {
                int cidx = tid + it * 256;
                int k    = cidx >> 5;          // 0..63
                int c4   = (cidx & 31) << 2;   // 0..124 (n in tile)
                unsigned short o[4];
                load4_bf(B + (size_t)(kk + k) * ldb + n0 + c4, o);
                #pragma unroll
                for (int j = 0; j < 4; ++j) Bs[(c4 + j) * SA + k] = o[j];
            }
        }
        __syncthreads();

        // ---- prefetch next K-stage into cache while we compute this one
        if (kk + BK < K) {
            int r = tid & 127;
            if (tid < 128) {
                __builtin_prefetch(A + (size_t)(m0 + r) * lda + kk + BK, 0, 0);
            } else if (B_PRETRANS) {
                __builtin_prefetch(B + (size_t)(n0 + r) * ldb + kk + BK, 0, 0);
            } else {
                __builtin_prefetch(B + (size_t)(kk + BK + (r >> 1)) * ldb + n0 + (r & 1) * 64, 0, 0);
            }
        }

        // ---- fragments (ISA 16-bit A 16x32 / B 32x16 lane striping)
        union Frag { uint4 u[2]; v16bf v; };
        #pragma unroll
        for (int ks = 0; ks < 2; ++ks) {       // two K-slices per LDS stage
            Frag a[2], b[4];
            #pragma unroll
            for (int mt = 0; mt < 2; ++mt) {
                int row = wm * 32 + mt * 16 + l15;
                const unsigned short* p = &As[row * SA + ks * 32 + hf * 8];
                a[mt].u[0] = *(const uint4*)p;          // K = hf*8   .. +7
                a[mt].u[1] = *(const uint4*)(p + 16);   // K = hf*8+16.. +23
            }
            #pragma unroll
            for (int nt = 0; nt < 4; ++nt) {
                int col = wn * 64 + nt * 16 + l15;
                const unsigned short* p = &Bs[col * SA + ks * 32 + hf * 8];
                b[nt].u[0] = *(const uint4*)p;          // K = hf*8   .. +7
                b[nt].u[1] = *(const uint4*)(p + 16);   // K = hf*8+16.. +23
            }
            #pragma unroll
            for (int mt = 0; mt < 2; ++mt)
                #pragma unroll
                for (int nt = 0; nt < 4; ++nt)
                    acc[mt][nt] = __builtin_amdgcn_wmma_f32_16x16x32_bf16(
                        false, a[mt].v, false, b[nt].v, (short)0, acc[mt][nt],
                        false, false);
        }
    }

    // ---- epilogue: C/D layout: VGPR r, lanes0-15 -> M=r, lanes16-31 -> M=r+8
    #pragma unroll
    for (int nt = 0; nt < 4; ++nt) {
        int col = n0 + wn * 64 + nt * 16 + l15;
        float bv = 0.0f;
        if (HAS_BIAS) bv = bias[col];
        #pragma unroll
        for (int mt = 0; mt < 2; ++mt) {
            int rbase = m0 + wm * 32 + mt * 16 + hf * 8;
            #pragma unroll
            for (int r = 0; r < 8; ++r)
                C[(size_t)(rbase + r) * ldc + col] = acc[mt][nt][r] + bv;
        }
    }
}

// ---------------------------------------------------------------------------
// LayerNorm over last dim D=1024, output bf16, optional extra scale (1/sqrt(D))
// ---------------------------------------------------------------------------
__global__ __launch_bounds__(256) void ln_bf16_kernel(
    const float* __restrict__ x, const float* __restrict__ g,
    const float* __restrict__ b, unsigned short* __restrict__ out, float scale)
{
    const int row = blockIdx.x;
    const int tid = threadIdx.x;
    const float* xr = x + (size_t)row * 1024;
    float v[4], s1 = 0.f, s2 = 0.f;
    #pragma unroll
    for (int i = 0; i < 4; ++i) {
        v[i] = xr[tid + i * 256];
        s1 += v[i]; s2 += v[i] * v[i];
    }
    __shared__ float r1[256], r2[256];
    r1[tid] = s1; r2[tid] = s2;
    __syncthreads();
    for (int off = 128; off > 0; off >>= 1) {
        if (tid < off) { r1[tid] += r1[tid + off]; r2[tid] += r2[tid + off]; }
        __syncthreads();
    }
    const float mean = r1[0] * (1.0f / 1024.0f);
    const float var  = r2[0] * (1.0f / 1024.0f) - mean * mean;
    const float rstd = rsqrtf(var + 1e-6f);
    unsigned short* orow = out + (size_t)row * 1024;
    #pragma unroll
    for (int i = 0; i < 4; ++i) {
        int c = tid + i * 256;
        orow[c] = f2bf(((v[i] - mean) * rstd * g[c] + b[c]) * scale);
    }
}

// ---------------------------------------------------------------------------
// Row softmax over 2048 cols; writes bf16 P, optionally f32 probs (a2 output)
// ---------------------------------------------------------------------------
__global__ __launch_bounds__(256) void softmax_kernel(
    const float* __restrict__ S, unsigned short* __restrict__ P,
    float* __restrict__ a2, int write_a2)
{
    const int row = blockIdx.x;
    const int tid = threadIdx.x;
    const float* sr = S + (size_t)row * 2048;
    float v[8], mx = -3.4e38f;
    #pragma unroll
    for (int i = 0; i < 8; ++i) { v[i] = sr[tid + i * 256]; mx = fmaxf(mx, v[i]); }
    __shared__ float red[256];
    red[tid] = mx;
    __syncthreads();
    for (int off = 128; off > 0; off >>= 1) {
        if (tid < off) red[tid] = fmaxf(red[tid], red[tid + off]);
        __syncthreads();
    }
    mx = red[0];
    __syncthreads();
    float s = 0.f;
    #pragma unroll
    for (int i = 0; i < 8; ++i) { v[i] = expf(v[i] - mx); s += v[i]; }
    red[tid] = s;
    __syncthreads();
    for (int off = 128; off > 0; off >>= 1) {
        if (tid < off) red[tid] += red[tid + off];
        __syncthreads();
    }
    const float inv = 1.0f / red[0];
    unsigned short* pr = P + (size_t)row * 2048;
    #pragma unroll
    for (int i = 0; i < 8; ++i) {
        int c = tid + i * 256;
        float p = v[i] * inv;
        pr[c] = f2bf(p);
        if (write_a2) a2[(size_t)row * 2048 + c] = p;
    }
}

// ---------------------------------------------------------------------------
// Residual concat: dst[row*2048 + off + col] = src[row*1024 + col]
// ---------------------------------------------------------------------------
__global__ __launch_bounds__(256) void copy_res_kernel(
    const float* __restrict__ src, float* __restrict__ dst, int off)
{
    size_t i   = (size_t)blockIdx.x * 256 + threadIdx.x;
    size_t row = i >> 10;
    int    col = (int)(i & 1023);
    dst[row * 2048 + off + col] = src[i];
}

// ---------------------------------------------------------------------------
extern "C" void kernel_launch(void* const* d_in, const int* in_sizes, int n_in,
                              void* d_out, int out_size, void* d_ws, size_t ws_size,
                              hipStream_t stream)
{
    (void)in_sizes; (void)n_in; (void)out_size; (void)ws_size;

    const float* q   = (const float*)d_in[0];
    const float* k   = (const float*)d_in[1];
    const float* v   = (const float*)d_in[2];
    const float* Wq1 = (const float*)d_in[3];  const float* bq1 = (const float*)d_in[4];
    const float* Wk1 = (const float*)d_in[5];  const float* bk1 = (const float*)d_in[6];
    const float* Wv1 = (const float*)d_in[7];  const float* bv1 = (const float*)d_in[8];
    const float* Wf1 = (const float*)d_in[9];  const float* bf1 = (const float*)d_in[10];
    const float* Wq2 = (const float*)d_in[11]; const float* bq2 = (const float*)d_in[12];
    const float* Wk2 = (const float*)d_in[13]; const float* bk2 = (const float*)d_in[14];
    const float* Wv2 = (const float*)d_in[15]; const float* bv2 = (const float*)d_in[16];
    const float* Wf2 = (const float*)d_in[17]; const float* bf2 = (const float*)d_in[18];
    const float* gq1 = (const float*)d_in[19]; const float* Bq1 = (const float*)d_in[20];
    const float* gk1 = (const float*)d_in[21]; const float* Bk1 = (const float*)d_in[22];
    const float* gv1 = (const float*)d_in[23]; const float* Bv1 = (const float*)d_in[24];
    const float* gq2 = (const float*)d_in[25]; const float* Bq2 = (const float*)d_in[26];
    const float* gk2 = (const float*)d_in[27]; const float* Bk2 = (const float*)d_in[28];
    const float* gv2 = (const float*)d_in[29]; const float* Bv2 = (const float*)d_in[30];

    // B=4, L=2048, D=1024 ; M = B*L = 8192
    const size_t ML = 8192ull * 1024ull;           // 8,388,608 elems

    char* ws = (char*)d_ws;
    float*          X1  = (float*)(ws);                       // 33.5 MB
    float*          X2  = (float*)(ws + 33554432u);
    float*          X3  = (float*)(ws + 67108864u);
    float*          O   = (float*)(ws + 100663296u);
    unsigned short* LNQ = (unsigned short*)(ws + 134217728u); // 16.8 MB
    unsigned short* LNK = (unsigned short*)(ws + 150994944u);
    unsigned short* LNV = (unsigned short*)(ws + 167772160u);
    unsigned short* P   = (unsigned short*)(ws + 184549376u); // 33.5 MB
    float*          S   = (float*)(ws + 218103808u);          // 67.1 MB

    float* outp = (float*)d_out;
    float* resp = outp + ML;            // (B,L,1,2D)
    float* a2p  = resp + 2ull * ML;     // (B,L,L)

    const dim3 blk(256);
    const dim3 gProj(8, 64, 1);         // N=1024, M=8192
    const dim3 gScore(16, 16, 4);       // N=2048, M=2048, batch 4
    const dim3 gPV(8, 16, 4);           // N=1024, M=2048, batch 4
    const float invT = 1.0f / 32.0f;    // 1/sqrt(1024)
    const size_t sQ = 2048ull * 1024ull, sS = 2048ull * 2048ull;

    // ---------------- layer 1 ----------------
    gemm_wmma<float, float, false, true><<<gProj, blk, 0, stream>>>(
        q, Wq1, bq1, X1, 8192, 1024, 1024, 1024, 1024, 1024, 0, 0, 0);
    gemm_wmma<float, float, false, true><<<gProj, blk, 0, stream>>>(
        k, Wk1, bk1, X2, 8192, 1024, 1024, 1024, 1024, 1024, 0, 0, 0);
    gemm_wmma<float, float, false, true><<<gProj, blk, 0, stream>>>(
        v, Wv1, bv1, X3, 8192, 1024, 1024, 1024, 1024, 1024, 0, 0, 0);

    copy_res_kernel<<<32768, blk, 0, stream>>>(X1, resp, 0);       // residual1

    ln_bf16_kernel<<<8192, blk, 0, stream>>>(X1, gq1, Bq1, LNQ, invT);
    ln_bf16_kernel<<<8192, blk, 0, stream>>>(X2, gk1, Bk1, LNK, 1.0f);
    ln_bf16_kernel<<<8192, blk, 0, stream>>>(X3, gv1, Bv1, LNV, 1.0f);

    gemm_wmma<unsigned short, unsigned short, true, false><<<gScore, blk, 0, stream>>>(
        LNQ, LNK, nullptr, S, 2048, 2048, 1024, 1024, 1024, 2048, sQ, sQ, sS);
    softmax_kernel<<<8192, blk, 0, stream>>>(S, P, nullptr, 0);
    gemm_wmma<unsigned short, unsigned short, false, false><<<gPV, blk, 0, stream>>>(
        P, LNV, nullptr, O, 2048, 1024, 2048, 2048, 1024, 1024, sS, sQ, sQ);

    gemm_wmma<float, float, false, true><<<gProj, blk, 0, stream>>>(
        O, Wf1, bf1, X1, 8192, 1024, 1024, 1024, 1024, 1024, 0, 0, 0);   // o1f -> X1

    // ---------------- layer 2 ----------------
    gemm_wmma<float, float, false, true><<<gProj, blk, 0, stream>>>(
        X1, Wq2, bq2, O, 8192, 1024, 1024, 1024, 1024, 1024, 0, 0, 0);   // q2 -> O
    copy_res_kernel<<<32768, blk, 0, stream>>>(O, resp, 1024);           // residual2
    gemm_wmma<float, float, false, true><<<gProj, blk, 0, stream>>>(
        k, Wk2, bk2, X2, 8192, 1024, 1024, 1024, 1024, 1024, 0, 0, 0);
    gemm_wmma<float, float, false, true><<<gProj, blk, 0, stream>>>(
        v, Wv2, bv2, X3, 8192, 1024, 1024, 1024, 1024, 1024, 0, 0, 0);

    ln_bf16_kernel<<<8192, blk, 0, stream>>>(O,  gq2, Bq2, LNQ, invT);
    ln_bf16_kernel<<<8192, blk, 0, stream>>>(X2, gk2, Bk2, LNK, 1.0f);
    ln_bf16_kernel<<<8192, blk, 0, stream>>>(X3, gv2, Bv2, LNV, 1.0f);

    gemm_wmma<unsigned short, unsigned short, true, false><<<gScore, blk, 0, stream>>>(
        LNQ, LNK, nullptr, S, 2048, 2048, 1024, 1024, 1024, 2048, sQ, sQ, sS);
    softmax_kernel<<<8192, blk, 0, stream>>>(S, P, a2p, 1);              // a2 out
    gemm_wmma<unsigned short, unsigned short, false, false><<<gPV, blk, 0, stream>>>(
        P, LNV, nullptr, X1, 2048, 1024, 2048, 2048, 1024, 1024, sS, sQ, sQ);

    gemm_wmma<float, float, false, true><<<gProj, blk, 0, stream>>>(
        X1, Wf2, bf2, outp, 8192, 1024, 1024, 1024, 1024, 1024, 0, 0, 0); // final out
}
